// KANLinear_87806311399518
// MI455X (gfx1250) — matrix operand order repaired and checked by reading, not compiled, not run on hardware
//
#include <hip/hip_runtime.h>

// ---------------------------------------------------------------------------
// KANLinear forward as ONE fused bf16 WMMA GEMM over augmented K = 6144:
//   out[n,o] = sum_j A[n,j]*W[o,j],  j = i*6 + r  (r=0: gelu(x), r=1..5: bases)
//
// Path A (ws >= 214MB): materialize A (bf16) once -> pure GEMM with
//   double-buffered global_load_async_to_lds_b128 pipeline (ASYNCcnt).
// Path B (small ws): fused GEMM computing A tiles on the fly (closed-form
//   quadratic B-splines instead of Cox-de-Boor: ~3x fewer VALU ops).
// Both paths: v_wmma_f32_16x16x32_bf16, f32 accumulation.
// ---------------------------------------------------------------------------

typedef __attribute__((ext_vector_type(16))) __bf16    v16bf;
typedef __attribute__((ext_vector_type(8)))  float     v8f;
typedef __attribute__((ext_vector_type(4)))  unsigned  u32x4;

#define IN_F   1024
#define OUT_F  1024
#define KAUG   6144          // 1024 * 6
#define M_TILE 128
#define N_TILE 128
#define KT     96            // 16 inputs * 6 cols = 3 WMMA k-steps
#define KI     16            // inputs per K chunk
#define LDSS   104           // padded LDS stride (halves): 16B-aligned, conflict-free
#define NCHUNK (KAUG / KT)   // 64

__device__ __forceinline__ unsigned short f2bf(float f) {
    unsigned u = __builtin_bit_cast(unsigned, f);
    u += 0x7FFFu + ((u >> 16) & 1u);      // round-to-nearest-even
    return (unsigned short)(u >> 16);
}

__device__ __forceinline__ float gelu_exact(float x) {
    return 0.5f * x * (1.0f + erff(x * 0.70710678118654752f));
}

// Quadratic B-spline bases on the reference grid g[j] = (j-2)*(2/3) - 1.
// Only 3 bases are nonzero: with s=(x+1)/h, c=floor(s), u=s-c:
//   B_c = (1-u)^2/2, B_{c+1} = 0.5+u(1-u), B_{c+2} = u^2/2  (others 0),
// truncated to output indices 0..4 (identical to the Cox-de-Boor reference).
__device__ __forceinline__ void kan_bases(float xv, float b[5]) {
    const float s  = __builtin_fmaf(xv, 1.5f, 1.5f);   // (x+1)/h, h = 2/3
    const float cf = floorf(s);
    const float u  = s - cf;
    const int   c  = (int)cf;
    const float n2  = 0.5f * u * u;
    const float omu = 1.0f - u;
    const float n0  = 0.5f * omu * omu;
    const float n1  = 1.0f - n0 - n2;                  // partition of unity
#pragma unroll
    for (int j = 0; j < 5; ++j) {
        const int d = j - c;
        b[j] = (d == 0) ? n0 : (d == 1) ? n1 : (d == 2) ? n2 : 0.0f;
    }
}

union BF16Frag { v16bf v; u32x4 q[2]; };

// --- CDNA5 async global->LDS (ASYNCcnt) ------------------------------------
__device__ __forceinline__ void async_ld_b128(unsigned lds_off, const void* gptr) {
    asm volatile("global_load_async_to_lds_b128 %0, %1, off"
                 :: "v"(lds_off), "v"(gptr) : "memory");
}
__device__ __forceinline__ void wait_async0() {
    asm volatile("s_wait_asynccnt 0" ::: "memory");
}

// ---------------------------------------------------------------------------
// Kernel 1: augmented weights W[o, i*6+0]=base, W[o, i*6+1+r]=spline*scaler
// ---------------------------------------------------------------------------
__global__ __launch_bounds__(256)
void kan_prep_w(const float* __restrict__ bw, const float* __restrict__ sw,
                const float* __restrict__ sc, unsigned short* __restrict__ Wbf) {
    const int idx = blockIdx.x * 256 + threadIdx.x;   // (o,i), 1024*1024 total
    const int o = idx >> 10;
    const int i = idx & 1023;
    const float b = bw[idx];
    const float s = sc[idx];
    const float* w = sw + (long)idx * 5;
    float v0 = w[0]*s, v1 = w[1]*s, v2 = w[2]*s, v3 = w[3]*s, v4 = w[4]*s;
    unsigned* d = (unsigned*)(Wbf + (long)o * KAUG + i * 6);
    d[0] = (unsigned)f2bf(b)  | ((unsigned)f2bf(v0) << 16);
    d[1] = (unsigned)f2bf(v1) | ((unsigned)f2bf(v2) << 16);
    d[2] = (unsigned)f2bf(v3) | ((unsigned)f2bf(v4) << 16);
}

// ---------------------------------------------------------------------------
// Kernel 2 (Path A): augmented activations A[n, i*6+r] in bf16
// ---------------------------------------------------------------------------
__global__ __launch_bounds__(256)
void kan_prep_a(const float* __restrict__ x, unsigned short* __restrict__ Abf) {
    const long idx = (long)blockIdx.x * 256 + threadIdx.x;   // n*1024 + i
    const float xv = x[idx];
    const float ge = gelu_exact(xv);
    float b[5];
    kan_bases(xv, b);
    unsigned* d = (unsigned*)(Abf + idx * 6);   // n*KAUG + i*6 == idx*6
    d[0] = (unsigned)f2bf(ge)   | ((unsigned)f2bf(b[0]) << 16);
    d[1] = (unsigned)f2bf(b[1]) | ((unsigned)f2bf(b[2]) << 16);
    d[2] = (unsigned)f2bf(b[3]) | ((unsigned)f2bf(b[4]) << 16);
}

// ---------------------------------------------------------------------------
// Kernel 3 (Path A): pure bf16 GEMM, double-buffered async-to-LDS pipeline
// ---------------------------------------------------------------------------
__global__ __launch_bounds__(256)
void kan_gemm_mat(const unsigned short* __restrict__ Abf,
                  const unsigned short* __restrict__ Wbf,
                  float* __restrict__ out) {
    __shared__ unsigned short Als[2][M_TILE * LDSS];   // 2 x 26 KB
    __shared__ unsigned short Bls[2][N_TILE * LDSS];   // 2 x 26 KB

    const int n0   = blockIdx.x * N_TILE;
    const int m0   = blockIdx.y * M_TILE;
    const int t    = threadIdx.x;
    const int lane = t & 31;
    const int wid  = t >> 5;
    const int wm   = (wid & 3) * 32;
    const int wn   = (wid >> 2) * 64;
    const int l16  = lane & 15;
    const int half = lane >> 4;

    // per-thread copy slot: 24 halves (3 x b128) at half-offset 0 or 48 of one tile row
    const int row  = t >> 1;
    const int part = (t & 1) * 48;
    const unsigned short* aSrc = Abf + (long)(m0 + row) * KAUG + part;
    const unsigned short* bSrc = Wbf + (long)(n0 + row) * KAUG + part;
    unsigned aDst[2], bDst[2];
#pragma unroll
    for (int bu = 0; bu < 2; ++bu) {
        aDst[bu] = (unsigned)(unsigned long long)&Als[bu][row * LDSS + part];
        bDst[bu] = (unsigned)(unsigned long long)&Bls[bu][row * LDSS + part];
    }

    v8f acc[2][4];
#pragma unroll
    for (int mi = 0; mi < 2; ++mi)
#pragma unroll
        for (int ni = 0; ni < 4; ++ni)
            acc[mi][ni] = (v8f){0.f,0.f,0.f,0.f,0.f,0.f,0.f,0.f};

    // prologue: async-load chunk 0 into buffer 0
#pragma unroll
    for (int q = 0; q < 3; ++q) {
        async_ld_b128(aDst[0] + q * 16, aSrc + q * 8);
        async_ld_b128(bDst[0] + q * 16, bSrc + q * 8);
    }

    for (int c = 0; c < NCHUNK; ++c) {
        const int cur = c & 1;
        wait_async0();        // chunk c resident in LDS (own lanes' share)
        __syncthreads();      // everyone's share resident

        if (c + 1 < NCHUNK) { // pipeline: fetch chunk c+1 into other buffer
            const unsigned short* ga = aSrc + (long)(c + 1) * KT;
            const unsigned short* gb = bSrc + (long)(c + 1) * KT;
            const int nxt = cur ^ 1;
#pragma unroll
            for (int q = 0; q < 3; ++q) {
                async_ld_b128(aDst[nxt] + q * 16, ga + q * 8);
                async_ld_b128(bDst[nxt] + q * 16, gb + q * 8);
            }
        }

#pragma unroll
        for (int ks = 0; ks < KT; ks += 32) {
            BF16Frag a[2], b[4];
#pragma unroll
            for (int mi = 0; mi < 2; ++mi) {
                // 16-bit A 16x32: lanes 0-15 K{0-7,16-23}, lanes 16-31 K{8-15,24-31}
                const unsigned short* p = &Als[cur][(wm + mi*16 + l16) * LDSS + ks];
                a[mi].q[0] = *(const u32x4*)(p + half * 8);
                a[mi].q[1] = *(const u32x4*)(p + 16 + half * 8);
            }
#pragma unroll
            for (int ni = 0; ni < 4; ++ni) {
                // 16-bit B 32x16: column = lane%16; lanes 0-15 K{0-7,16-23}, lanes 16-31 K{8-15,24-31}
                const unsigned short* p = &Bls[cur][(wn + ni*16 + l16) * LDSS + ks];
                b[ni].q[0] = *(const u32x4*)(p + half * 8);
                b[ni].q[1] = *(const u32x4*)(p + 16 + half * 8);
            }
#pragma unroll
            for (int mi = 0; mi < 2; ++mi)
#pragma unroll
                for (int ni = 0; ni < 4; ++ni)
                    acc[mi][ni] = __builtin_amdgcn_wmma_f32_16x16x32_bf16(
                        false, a[mi].v, false, b[ni].v,
                        (short)0, acc[mi][ni], false, false);
        }
        __syncthreads();      // all reads of buffer `cur` done before reuse
    }

    // C/D layout: VGPR v holds M=v (lanes 0-15) / M=v+8 (lanes 16-31), N=lane%16
#pragma unroll
    for (int mi = 0; mi < 2; ++mi)
#pragma unroll
        for (int ni = 0; ni < 4; ++ni)
#pragma unroll
            for (int v = 0; v < 8; ++v) {
                const int r = m0 + wm + mi*16 + v + half*8;
                const int cN = n0 + wn + ni*16 + l16;
                out[(long)r * OUT_F + cN] = acc[mi][ni][v];
            }
}

// ---------------------------------------------------------------------------
// Kernel (Path B fallback): fused GEMM, A tile generated on the fly
// ---------------------------------------------------------------------------
__global__ __launch_bounds__(256)
void kan_gemm_fused(const float* __restrict__ x,
                    const unsigned short* __restrict__ Wbf,
                    float* __restrict__ out) {
    __shared__ unsigned short Als[M_TILE * LDSS];
    __shared__ unsigned short Bls[N_TILE * LDSS];

    const int n0   = blockIdx.x * N_TILE;
    const int m0   = blockIdx.y * M_TILE;
    const int t    = threadIdx.x;
    const int lane = t & 31;
    const int wid  = t >> 5;
    const int wm   = (wid & 3) * 32;
    const int wn   = (wid >> 2) * 64;
    const int l16  = lane & 15;
    const int half = lane >> 4;

    v8f acc[2][4];
#pragma unroll
    for (int mi = 0; mi < 2; ++mi)
#pragma unroll
        for (int ni = 0; ni < 4; ++ni)
            acc[mi][ni] = (v8f){0.f,0.f,0.f,0.f,0.f,0.f,0.f,0.f};

    for (int chunk = 0; chunk < NCHUNK; ++chunk) {
        const int i0 = chunk * KI;
        const int k0 = chunk * KT;

        { // A tile: gelu + closed-form bases -> bf16 LDS
            const int ci    = t & 15;
            const int rbase = t >> 4;
            const int ii    = i0 + ci;
#pragma unroll
            for (int rep = 0; rep < 8; ++rep) {
                const int mrow = rbase + rep * 16;
                const float xv = x[(long)(m0 + mrow) * IN_F + ii];
                const float ge = gelu_exact(xv);
                float b2[5];
                kan_bases(xv, b2);
                unsigned* d = (unsigned*)&Als[mrow * LDSS + ci * 6];
                d[0] = (unsigned)f2bf(ge)    | ((unsigned)f2bf(b2[0]) << 16);
                d[1] = (unsigned)f2bf(b2[1]) | ((unsigned)f2bf(b2[2]) << 16);
                d[2] = (unsigned)f2bf(b2[3]) | ((unsigned)f2bf(b2[4]) << 16);
            }
        }
        { // B tile
            const int row  = t >> 1;
            const int part = (t & 1) * 48;
            const unsigned short* src = Wbf + (long)(n0 + row) * KAUG + k0 + part;
            u32x4* dst = (u32x4*)&Bls[row * LDSS + part];
#pragma unroll
            for (int q = 0; q < 3; ++q) dst[q] = ((const u32x4*)src)[q];
            __builtin_prefetch(src + KT, 0, 0);
        }
        __syncthreads();

#pragma unroll
        for (int ks = 0; ks < KT; ks += 32) {
            BF16Frag a[2], b[4];
#pragma unroll
            for (int mi = 0; mi < 2; ++mi) {
                const unsigned short* p = &Als[(wm + mi*16 + l16) * LDSS + ks];
                a[mi].q[0] = *(const u32x4*)(p + half * 8);
                a[mi].q[1] = *(const u32x4*)(p + 16 + half * 8);
            }
#pragma unroll
            for (int ni = 0; ni < 4; ++ni) {
                const unsigned short* p = &Bls[(wn + ni*16 + l16) * LDSS + ks];
                b[ni].q[0] = *(const u32x4*)(p + half * 8);
                b[ni].q[1] = *(const u32x4*)(p + 16 + half * 8);
            }
#pragma unroll
            for (int mi = 0; mi < 2; ++mi)
#pragma unroll
                for (int ni = 0; ni < 4; ++ni)
                    acc[mi][ni] = __builtin_amdgcn_wmma_f32_16x16x32_bf16(
                        false, a[mi].v, false, b[ni].v,
                        (short)0, acc[mi][ni], false, false);
        }
        __syncthreads();
    }

#pragma unroll
    for (int mi = 0; mi < 2; ++mi)
#pragma unroll
        for (int ni = 0; ni < 4; ++ni)
#pragma unroll
            for (int v = 0; v < 8; ++v) {
                const int r = m0 + wm + mi*16 + v + half*8;
                const int cN = n0 + wn + ni*16 + l16;
                out[(long)r * OUT_F + cN] = acc[mi][ni][v];
            }
}

// ---------------------------------------------------------------------------
extern "C" void kernel_launch(void* const* d_in, const int* in_sizes, int n_in,
                              void* d_out, int out_size, void* d_ws, size_t ws_size,
                              hipStream_t stream) {
    const float* x  = (const float*)d_in[0];   // (N, 1024)
    const float* bw = (const float*)d_in[1];   // (1024, 1024)
    const float* sw = (const float*)d_in[2];   // (1024, 1024, 5)
    const float* sc = (const float*)d_in[3];   // (1024, 1024)
    float* out = (float*)d_out;

    const int    Nrows  = in_sizes[0] / IN_F;                 // 16384
    const size_t ABYTES = (size_t)Nrows * KAUG * 2;           // ~201 MB
    const size_t WBYTES = (size_t)OUT_F * KAUG * 2;           // ~12.6 MB
    dim3 grid(OUT_F / N_TILE, Nrows / M_TILE);                // (8, 128)

    if (ws_size >= ABYTES + WBYTES) {
        // Path A: materialize A once, pipelined async GEMM
        unsigned short* Abf = (unsigned short*)d_ws;
        unsigned short* Wbf = (unsigned short*)((char*)d_ws + ABYTES);
        kan_prep_w<<<(OUT_F * IN_F) / 256, 256, 0, stream>>>(bw, sw, sc, Wbf);
        kan_prep_a<<<((long)Nrows * IN_F) / 256, 256, 0, stream>>>(x, Abf);
        kan_gemm_mat<<<grid, 256, 0, stream>>>(Abf, Wbf, out);
    } else {
        // Path B: fused activation + GEMM (only 12.6 MB workspace)
        unsigned short* Wbf = (unsigned short*)d_ws;
        kan_prep_w<<<(OUT_F * IN_F) / 256, 256, 0, stream>>>(bw, sw, sc, Wbf);
        kan_gemm_fused<<<grid, 256, 0, stream>>>(x, Wbf, out);
    }
}
